// MultiHeadAttention_87780541595793
// MI455X (gfx1250) — matrix operand order repaired and checked by reading, not compiled, not run on hardware
//
#include <hip/hip_runtime.h>

#define DIN    1024
#define DOUT   1024
#define NHEAD  16
#define DHD    64
#define NBATCH 4
#define SEQ    2048

typedef _Float16 v16h __attribute__((ext_vector_type(16)));
typedef _Float16 v8h  __attribute__((ext_vector_type(8)));
typedef _Float16 v4h  __attribute__((ext_vector_type(4)));
typedef float    v8f  __attribute__((ext_vector_type(8)));
typedef int      v4i  __attribute__((ext_vector_type(4)));

union AFrag { v16h v; v8h h[2]; };

static __device__ __forceinline__ v8f wmma16(v16h a, v16h b, v8f c) {
  return __builtin_amdgcn_wmma_f32_16x16x32_f16(false, a, false, b, (short)0, c,
                                                false, false);
}

// ---- CDNA5 async global->LDS copy (ASYNCcnt path) --------------------------
// Probe confirmed: signature takes (v4i AS1*, v4i AS3*, imm offset, imm cpol).
#if __has_builtin(__builtin_amdgcn_global_load_async_to_lds_b128)
#define HAVE_ASYNC 1
static __device__ __forceinline__ void async_cp16(void* lds, const void* g) {
  __builtin_amdgcn_global_load_async_to_lds_b128(
      (__attribute__((address_space(1))) v4i*)g,
      (__attribute__((address_space(3))) v4i*)lds, 0, 0);
}
static __device__ __forceinline__ void wait_async0() {
#if __has_builtin(__builtin_amdgcn_s_wait_asynccnt)
  __builtin_amdgcn_s_wait_asynccnt(0);
#else
  asm volatile("s_wait_asynccnt 0" ::: "memory");
#endif
}
#else
#define HAVE_ASYNC 0
#endif

// ---------------------------------------------------------------------------
// One-time f32 -> f16 convert of x  (8M elements, pure stream)
// ---------------------------------------------------------------------------
__global__ __launch_bounds__(256) void conv_x_kernel(
    const float* __restrict__ X, _Float16* __restrict__ Xh)
{
  const int idx = blockIdx.x * 256 + threadIdx.x;   // 524288 threads, 4 vec4 each
  #pragma unroll
  for (int it = 0; it < 4; ++it) {
    const size_t i = (size_t)(idx + it * 524288) * 4;
    float4 f = *(const float4*)(X + i);
    v4h h = { (_Float16)f.x, (_Float16)f.y, (_Float16)f.z, (_Float16)f.w };
    *(v4h*)(Xh + i) = h;
  }
}

// ---------------------------------------------------------------------------
// One-time W convert + transpose: W[k][n] f32 -> Wt[n][k] f16 (per z = q/k/v)
// ---------------------------------------------------------------------------
__global__ __launch_bounds__(256) void conv_w_kernel(
    const float* __restrict__ Wq, const float* __restrict__ Wk,
    const float* __restrict__ Wv, _Float16* __restrict__ Wt)
{
  __shared__ _Float16 Ts[64][72];                  // [n][k], padded rows
  const int tid = threadIdx.x;
  const int k0 = blockIdx.x * 64, n0 = blockIdx.y * 64;
  const float* W = (blockIdx.z == 0) ? Wq : (blockIdx.z == 1) ? Wk : Wv;
  _Float16* Wz = Wt + (size_t)blockIdx.z * DIN * DOUT;
  #pragma unroll
  for (int it = 0; it < 4; ++it) {                 // 1024 float4 reads
    int i = tid + it * 256;
    int r = i >> 4, c = (i & 15) * 4;              // k-local, n-local
    float4 f = *(const float4*)(W + (size_t)(k0 + r) * DOUT + n0 + c);
    Ts[c + 0][r] = (_Float16)f.x; Ts[c + 1][r] = (_Float16)f.y;
    Ts[c + 2][r] = (_Float16)f.z; Ts[c + 3][r] = (_Float16)f.w;
  }
  __syncthreads();
  #pragma unroll
  for (int it = 0; it < 2; ++it) {                 // 512 v8h writes
    int i = tid + it * 256;
    int n = i >> 3, kk = (i & 7) * 8;
    *(v8h*)(Wz + (size_t)(n0 + n) * DIN + k0 + kk) = *(const v8h*)&Ts[n][kk];
  }
}

// ---------------------------------------------------------------------------
// QKV projection, all-f16 data path: C[8192x1024] = Xh * W  (per z)
// WG = 4 waves -> 64(M) x 128(N) tile; wave = 32x64 (8 WMMA / 32-K chunk).
// Double-buffered LDS; staging is a pure B128 copy (async when available).
// ---------------------------------------------------------------------------
__global__ __launch_bounds__(128) void qkv_gemm_kernel(
    const _Float16* __restrict__ Xh, const _Float16* __restrict__ Wt,
    _Float16* __restrict__ Qo, _Float16* __restrict__ Ko, _Float16* __restrict__ Vo)
{
  __shared__ _Float16 Xs[2][64][32];
  __shared__ _Float16 Ws[2][128][32];              // Wt tile: [n][k]
  const int tid  = threadIdx.x;
  const int wave = tid >> 5, lane = tid & 31;
  const int wm = wave >> 1, wn = wave & 1;
  const int m0 = blockIdx.x * 64;
  const int n0 = blockIdx.y * 128;
  const _Float16* Wz = Wt + (size_t)blockIdx.z * DIN * DOUT;
  _Float16* Out = (blockIdx.z == 0) ? Qo : (blockIdx.z == 1) ? Ko : Vo;

  const int koff = (lane >> 4) * 8;
  const int boff = (lane >> 4) * 16;
  v8f acc[2][4] = {};

  // Constant-trip staging indices (no divergent bounds checks)
  const int xr = tid >> 2, xc = (tid & 3) * 8;
#if HAVE_ASYNC
  auto stage = [&](int p, int kc) {
    const int k0 = kc * 32;
    #pragma unroll
    for (int it = 0; it < 2; ++it) {
      int r = xr + it * 32;
      async_cp16(&Xs[p][r][xc], Xh + (size_t)(m0 + r) * DIN + k0 + xc);
    }
    #pragma unroll
    for (int it = 0; it < 4; ++it) {
      int r = xr + it * 32;
      async_cp16(&Ws[p][r][xc], Wz + (size_t)(n0 + r) * DIN + k0 + xc);
    }
  };
#else
  uint4 regX[2], regW[4];
  auto loadregs = [&](int kc) {
    const int k0 = kc * 32;
    #pragma unroll
    for (int it = 0; it < 2; ++it)
      regX[it] = *(const uint4*)(Xh + (size_t)(m0 + xr + it * 32) * DIN + k0 + xc);
    #pragma unroll
    for (int it = 0; it < 4; ++it)
      regW[it] = *(const uint4*)(Wz + (size_t)(n0 + xr + it * 32) * DIN + k0 + xc);
  };
  auto storeregs = [&](int p) {
    #pragma unroll
    for (int it = 0; it < 2; ++it) *(uint4*)&Xs[p][xr + it * 32][xc] = regX[it];
    #pragma unroll
    for (int it = 0; it < 4; ++it) *(uint4*)&Ws[p][xr + it * 32][xc] = regW[it];
  };
#endif

  auto compute = [&](int p) {
    AFrag a[2];
    #pragma unroll
    for (int mi = 0; mi < 2; ++mi) {
      int row = wm * 32 + mi * 16 + (lane & 15);
      a[mi].h[0] = *(const v8h*)&Xs[p][row][boff];
      a[mi].h[1] = *(const v8h*)&Xs[p][row][boff + 8];
    }
    #pragma unroll
    for (int nb = 0; nb < 4; ++nb) {
      v16h b = *(const v16h*)&Ws[p][wn * 64 + nb * 16 + (lane & 15)][boff];
      acc[0][nb] = wmma16(a[0].v, b, acc[0][nb]);
      acc[1][nb] = wmma16(a[1].v, b, acc[1][nb]);
    }
  };

#if HAVE_ASYNC
  stage(0, 0);
  for (int kc = 0; kc < DIN / 32; ++kc) {
    const int p = kc & 1;
    wait_async0();
    __syncthreads();
    if (kc + 1 < DIN / 32) stage(p ^ 1, kc + 1);
    compute(p);
  }
#else
  loadregs(0);
  for (int kc = 0; kc < DIN / 32; ++kc) {
    const int p = kc & 1;
    storeregs(p);
    __syncthreads();
    if (kc + 1 < DIN / 32) loadregs(kc + 1);
    compute(p);
    __syncthreads();
  }
#endif

  #pragma unroll
  for (int mi = 0; mi < 2; ++mi) {
    #pragma unroll
    for (int nb = 0; nb < 4; ++nb) {
      #pragma unroll
      for (int r = 0; r < 8; ++r) {
        int M   = r + ((lane >> 4) << 3);
        int row = m0 + wm * 32 + mi * 16 + M;
        int col = n0 + wn * 64 + nb * 16 + (lane & 15);
        int bb = row >> 11, tt = row & (SEQ - 1);
        int hh = col >> 6,  dd = col & (DHD - 1);
        Out[(((size_t)(bb * NHEAD + hh)) * SEQ + tt) * DHD + dd] =
            (_Float16)acc[mi][nb][r];
      }
    }
  }
}

// ---------------------------------------------------------------------------
// Flash attention, causal. WG = 4 waves / (b*h, 64-query chunk); K/V stream
// through double-buffered LDS in 32-key blocks (K async when available).
// ---------------------------------------------------------------------------
__global__ __launch_bounds__(128) void flash_attn_kernel(
    const _Float16* __restrict__ Q, const _Float16* __restrict__ K,
    const _Float16* __restrict__ V, float* __restrict__ Out)
{
  __shared__ _Float16 Ks[2][32][64];               // [key][d]
  __shared__ _Float16 Vt[2][64][32];               // transposed [d][key]
  __shared__ _Float16 Ps[4][16][32];               // per-wave P relayout
  const int tid = threadIdx.x, wave = tid >> 5, lane = tid & 31;
  const int qc = blockIdx.x, bh = blockIdx.y;
  const _Float16* Qb = Q + (size_t)bh * SEQ * DHD;
  const _Float16* Kb = K + (size_t)bh * SEQ * DHD;
  const _Float16* Vb = V + (size_t)bh * SEQ * DHD;
  const int q0   = qc * 64 + wave * 16;
  const int koff = (lane >> 4) * 8;
  const int boff = (lane >> 4) * 16;
  const int cr = tid >> 3, cc = (tid & 7) * 8;     // staging coords (32x64 tile)

  AFrag qf[2];
  {
    const _Float16* qp = Qb + (size_t)(q0 + (lane & 15)) * DHD;
    #pragma unroll
    for (int c = 0; c < 2; ++c) {
      qf[c].h[0] = *(const v8h*)(qp + 32 * c + boff);
      qf[c].h[1] = *(const v8h*)(qp + 32 * c + boff + 8);
    }
  }
  v8f o[4] = {};
  float mr[8], lr[8];
  #pragma unroll
  for (int r = 0; r < 8; ++r) { mr[r] = -1e30f; lr[r] = 0.f; }

  v8h vreg[2];
#if !HAVE_ASYNC
  uint4 kreg[2];
#endif
  auto prefetch = [&](int kb) {                    // issue global loads
    const int kv0 = kb * 32;
    #pragma unroll
    for (int it = 0; it < 2; ++it) {
      int r = cr + it * 16;
      vreg[it] = *(const v8h*)(Vb + (size_t)(kv0 + r) * DHD + cc);
#if !HAVE_ASYNC
      kreg[it] = *(const uint4*)(Kb + (size_t)(kv0 + r) * DHD + cc);
#endif
    }
  };
#if HAVE_ASYNC
  auto stageK = [&](int p, int kb) {
    const int kv0 = kb * 32;
    #pragma unroll
    for (int it = 0; it < 2; ++it) {
      int r = cr + it * 16;
      async_cp16(&Ks[p][r][cc], Kb + (size_t)(kv0 + r) * DHD + cc);
    }
  };
#endif
  auto storeTiles = [&](int p) {                   // V transpose (+K fallback)
    #pragma unroll
    for (int it = 0; it < 2; ++it) {
      int r = cr + it * 16;
#if !HAVE_ASYNC
      *(uint4*)&Ks[p][r][cc] = kreg[it];
#endif
      #pragma unroll
      for (int j = 0; j < 8; ++j) Vt[p][cc + j][r] = vreg[it][j];
    }
  };

  const int nblk = 2 * qc + 2;                     // causal upper bound
#if HAVE_ASYNC
  stageK(0, 0);
#endif
  prefetch(0);
  for (int kb = 0; kb < nblk; ++kb) {
    const int p = kb & 1;
    storeTiles(p);
#if HAVE_ASYNC
    wait_async0();
#endif
    __syncthreads();
    if (kb + 1 < nblk) {
#if HAVE_ASYNC
      stageK(p ^ 1, kb + 1);
#endif
      prefetch(kb + 1);
    }

    // S = Q * K^T (two 16x16 frags over 32 keys, contraction d = 64)
    v8f s0 = {}, s1 = {};
    #pragma unroll
    for (int c = 0; c < 2; ++c) {
      v16h b0 = *(const v16h*)&Ks[p][(lane & 15)][32 * c + boff];
      s0 = wmma16(qf[c].v, b0, s0);
      v16h b1 = *(const v16h*)&Ks[p][16 + (lane & 15)][32 * c + boff];
      s1 = wmma16(qf[c].v, b1, s1);
    }

    // Online softmax in exp2 domain: scale = log2(e)/sqrt(64)
    const int kv0 = kb * 32;
    const float SC = 0.18033688011118312f;
    #pragma unroll
    for (int r = 0; r < 8; ++r) {
      float a0 = s0[r] * SC, a1 = s1[r] * SC;
      const int qrow = q0 + r + ((lane >> 4) << 3);
      const int key0 = kv0 + (lane & 15);
      const bool ok0 = key0 <= qrow, ok1 = key0 + 16 <= qrow;
      float m0v = ok0 ? a0 : -1e30f;
      float m1v = ok1 ? a1 : -1e30f;
      float mx = fmaxf(m0v, m1v);
      #pragma unroll
      for (int d = 1; d < 16; d <<= 1) mx = fmaxf(mx, __shfl_xor(mx, d));
      const float nm    = fmaxf(mr[r], mx);
      const float alpha = __builtin_amdgcn_exp2f(mr[r] - nm);
      float p0 = ok0 ? __builtin_amdgcn_exp2f(a0 - nm) : 0.f;
      float p1 = ok1 ? __builtin_amdgcn_exp2f(a1 - nm) : 0.f;
      float ls = p0 + p1;
      #pragma unroll
      for (int d = 1; d < 16; d <<= 1) ls += __shfl_xor(ls, d);
      lr[r] = lr[r] * alpha + ls;
      mr[r] = nm;
      #pragma unroll
      for (int dc = 0; dc < 4; ++dc) o[dc][r] *= alpha;
      const int M = r + ((lane >> 4) << 3);
      Ps[wave][M][lane & 15]        = (_Float16)p0;
      Ps[wave][M][16 + (lane & 15)] = (_Float16)p1;
    }
    asm volatile("s_wait_dscnt 0" ::: "memory");   // P stores -> P frag loads

    AFrag pf;
    pf.h[0] = *(const v8h*)&Ps[wave][lane & 15][boff];
    pf.h[1] = *(const v8h*)&Ps[wave][lane & 15][boff + 8];
    #pragma unroll
    for (int dc = 0; dc < 4; ++dc) {
      v16h vf = *(const v16h*)&Vt[p][dc * 16 + (lane & 15)][boff];
      o[dc] = wmma16(pf.v, vf, o[dc]);
    }
    __syncthreads();                               // tiles free for overwrite
  }

  const int bb = bh >> 4, hh = bh & 15;
  #pragma unroll
  for (int dc = 0; dc < 4; ++dc) {
    #pragma unroll
    for (int r = 0; r < 8; ++r) {
      int M  = r + ((lane >> 4) << 3);
      int tt = q0 + M;
      Out[((size_t)(bb * SEQ + tt)) * DOUT + hh * DHD + dc * 16 + (lane & 15)] =
          o[dc][r] / lr[r];
    }
  }
}

extern "C" void kernel_launch(void* const* d_in, const int* in_sizes, int n_in,
                              void* d_out, int out_size, void* d_ws, size_t ws_size,
                              hipStream_t stream) {
  (void)in_sizes; (void)n_in; (void)out_size; (void)ws_size;
  const float* x  = (const float*)d_in[0];
  const float* wq = (const float*)d_in[1];
  const float* wk = (const float*)d_in[2];
  const float* wv = (const float*)d_in[3];
  float* out = (float*)d_out;

  // Workspace layout (f16 elements)
  _Float16* Xh = (_Float16*)d_ws;                         //  8M (x in f16)
  _Float16* Wt = Xh + (size_t)NBATCH * SEQ * DIN;         //  3M (W^T f16, 3x)
  _Float16* Qh = Wt + (size_t)3 * DIN * DOUT;             //  8M
  _Float16* Kh = Qh + (size_t)NBATCH * NHEAD * SEQ * DHD; //  8M
  _Float16* Vh = Kh + (size_t)NBATCH * NHEAD * SEQ * DHD; //  8M

  conv_x_kernel<<<2048, 256, 0, stream>>>(x, Xh);
  conv_w_kernel<<<dim3(DIN / 64, DOUT / 64, 3), 256, 0, stream>>>(wq, wk, wv, Wt);
  qkv_gemm_kernel<<<dim3((NBATCH * SEQ) / 64, DOUT / 128, 3), 128, 0, stream>>>(
      Xh, Wt, Qh, Kh, Vh);
  flash_attn_kernel<<<dim3(SEQ / 64, NBATCH * NHEAD), 128, 0, stream>>>(
      Qh, Kh, Vh, out);
}
